// PLELevel_60722247631591
// MI455X (gfx1250) — compile-verified, matrix operand order checked by reading, not run on hardware
//
#include <hip/hip_runtime.h>
#include <math.h>

// ---------------------------------------------------------------- constants
static constexpr int T  = 4;
static constexpr int Bt = 4096;   // batch
static constexpr int D  = 512;
static constexpr int H  = 1024;
static constexpr int ES = 4;
static constexpr int ET = 4;
static constexpr int NEXP = ES + ET;      // 8
static constexpr float INV_SCALE = 0.04419417382415922f; // 1/sqrt(512)
static constexpr float EPS = 1e-5f;

typedef unsigned short u16;
typedef __attribute__((ext_vector_type(16))) __bf16   v16bf;
typedef __attribute__((ext_vector_type(8)))  float    v8f;
typedef __attribute__((ext_vector_type(8)))  unsigned int v8u;
typedef int v4i __attribute__((vector_size(16)));           // GCC-style <4 x i32>
typedef __attribute__((address_space(1))) v4i gv4i;         // global
typedef __attribute__((address_space(3))) v4i lv4i;         // LDS

// ------------------------------------------------- CDNA5 async-to-LDS probe
#if defined(__has_builtin)
#  if __has_builtin(__builtin_amdgcn_global_load_async_to_lds_b128) && \
      __has_builtin(__builtin_amdgcn_s_wait_asynccnt)
#    define PLE_ASYNC 1
#  endif
#endif
#ifndef PLE_ASYNC
#  define PLE_ASYNC 0
#endif

#if PLE_ASYNC
#  define PLE_WAIT8() __builtin_amdgcn_s_wait_asynccnt(8)
#  define PLE_WAIT0() __builtin_amdgcn_s_wait_asynccnt(0)
#else
#  define PLE_WAIT8()
#  define PLE_WAIT0()
#endif

// 16-byte global->LDS copy: async DMA (ASYNCcnt) when available, else b128.
__device__ __forceinline__ void cp16(const u16* g, u16* l) {
#if PLE_ASYNC
  __builtin_amdgcn_global_load_async_to_lds_b128((gv4i*)g, (lv4i*)l, 0, 0);
#else
  *(uint4*)l = *(const uint4*)g;
#endif
}

// ---------------------------------------------------------------- helpers
__device__ __forceinline__ u16 f2bf(float f) {
  unsigned u = __float_as_uint(f);
  unsigned r = u + 0x7FFFu + ((u >> 16) & 1u);   // round-to-nearest-even
  return (u16)(r >> 16);
}

__device__ __forceinline__ float wave_sum(float v) {
#pragma unroll
  for (int m = 16; m > 0; m >>= 1) v += __shfl_xor(v, m, 32);
  return v;
}

__device__ __forceinline__ float gelu_exact(float x) {
  return 0.5f * x * (1.0f + erff(x * 0.70710678118654752f));
}

// ---------------------------------------------------------------- prep
__global__ void __launch_bounds__(256) prep_x_kernel(
    const float* __restrict__ x, float* __restrict__ xs_f,
    u16* __restrict__ xs_b, u16* __restrict__ x_b) {
  int idx = blockIdx.x * 256 + threadIdx.x;          // over B*D
  float a0 = x[idx];
  float a1 = x[1 * Bt * D + idx];
  float a2 = x[2 * Bt * D + idx];
  float a3 = x[3 * Bt * D + idx];
  x_b[idx]              = f2bf(a0);
  x_b[1 * Bt * D + idx] = f2bf(a1);
  x_b[2 * Bt * D + idx] = f2bf(a2);
  x_b[3 * Bt * D + idx] = f2bf(a3);
  float m = (a0 + a1 + a2 + a3) * 0.25f;
  xs_f[idx] = m;
  xs_b[idx] = f2bf(m);
}

__global__ void __launch_bounds__(256) cvt_bf16_kernel(
    const float* __restrict__ in, u16* __restrict__ out) {
  int idx = blockIdx.x * 256 + threadIdx.x;
  out[idx] = f2bf(in[idx]);
}

// f32 [K,N] (batched, z) -> bf16 [N,K]  (LDS 32x32 tile transpose)
__global__ void __launch_bounds__(256) cvt_t_kernel(
    const float* __restrict__ in, u16* __restrict__ out, int K, int N) {
  __shared__ float tile[32][33];
  size_t base = (size_t)blockIdx.z * K * N;
  int tx = threadIdx.x & 31, ty = threadIdx.x >> 5;
#pragma unroll
  for (int i = 0; i < 32; i += 8) {
    int k = blockIdx.y * 32 + ty + i, n = blockIdx.x * 32 + tx;
    tile[ty + i][tx] = in[base + (size_t)k * N + n];
  }
  __syncthreads();
#pragma unroll
  for (int i = 0; i < 32; i += 8) {
    int n = blockIdx.x * 32 + ty + i, k = blockIdx.y * 32 + tx;
    out[base + (size_t)n * K + k] = f2bf(tile[tx][ty + i]);
  }
}

// cb[t][d] = sum_e qb[t][e] * kw[t][d][e]   (folded q-bias through kw^T)
__global__ void __launch_bounds__(256) cbias_kernel(
    const float* __restrict__ qb, const float* __restrict__ kw,
    float* __restrict__ cb) {
  int idx = blockIdx.x * 256 + threadIdx.x;          // over T*D
  int t = idx >> 9, d = idx & (D - 1);
  const float* qbr = qb + (size_t)t * D;
  const float* kwr = kw + ((size_t)t * D + d) * D;
  float s = 0.f;
  for (int e = 0; e < D; ++e) s += qbr[e] * kwr[e];
  cb[idx] = s;
}

// ---------------------------------------------------------------- WMMA GEMM
// C[M,N] = act(A @ B^T_stored + bias); A bf16 [M,K] row-major,
// Bm bf16 stored [N,K] row-major (logical B[k][n] = Bm[n*K+k]).
// 256 threads = 8 waves (4 along M x 2 along N), block tile 128x128,
// K-step 64 (2 x v_wmma_f32_16x16x32_bf16 per accumulator per stage),
// double-buffered LDS, XOR-swizzled 16B chunks, async global->LDS staging.

// LDS tile row = 64 bf16 = 128 B = 8 chunks of 16 B; chunk c stored at
// c ^ ((row>>1)&7) -> conflict-free fragment reads, 16B-aligned transfers.
__device__ __forceinline__ void stage_tiles(const u16* A, const u16* Bm,
                                            u16* bufA, u16* bufB, int m0,
                                            int n0, int k0, int K, int tid) {
#pragma unroll
  for (int j = 0; j < 4; ++j) {
    int row = 32 * j + (tid >> 3);
    int c = tid & 7;
    int sw = c ^ ((row >> 1) & 7);
    cp16(A + (size_t)(m0 + row) * K + k0 + c * 8, bufA + row * 64 + sw * 8);
    cp16(Bm + (size_t)(n0 + row) * K + k0 + c * 8, bufB + row * 64 + sw * 8);
  }
}

// ISA 16-bit fragment: lane half hi, K-pairs {0..7,16..23}(hi=0)/{8..15,24..31}
__device__ __forceinline__ v16bf load_frag(const u16* buf, int row, int ks,
                                           int hi) {
  v8u u;
#pragma unroll
  for (int p = 0; p < 8; ++p) {
    int off = ((p < 4) ? (4 * p) : (32 + 4 * (p - 4))) + 16 * hi + 64 * ks;
    int ch = off >> 4;
    int sw = ch ^ ((row >> 1) & 7);
    u[p] = *(const unsigned int*)(buf + row * 64 + sw * 8 + ((off & 15) >> 1));
  }
  return __builtin_bit_cast(v16bf, u);
}

template <bool GELU, bool OUT_BF16>
__global__ void __launch_bounds__(256) gemm_bf16_kernel(
    const u16* __restrict__ A, const u16* __restrict__ Bm,
    const float* __restrict__ bias, void* __restrict__ Cout,
    int M, int N, int K) {
  __shared__ u16 lA[2][128 * 64];
  __shared__ u16 lB[2][128 * 64];

  const int tid  = threadIdx.x;
  const int lane = tid & 31;
  const int r    = lane & 15;
  const int hi   = lane >> 4;
  const int wv   = tid >> 5;
  const int wm   = wv & 3;      // wave row (x32)
  const int wn   = wv >> 2;     // wave col (x64)

  const int m0 = blockIdx.y * 128;
  const int n0 = blockIdx.x * 128;

  v8f acc[2][4];
#pragma unroll
  for (int mt = 0; mt < 2; ++mt)
#pragma unroll
    for (int nt = 0; nt < 4; ++nt)
#pragma unroll
      for (int v = 0; v < 8; ++v) acc[mt][nt][v] = 0.0f;

  auto compute = [&](int kb) {
    const u16* bA = lA[kb & 1];
    const u16* bB = lB[kb & 1];
#pragma unroll
    for (int ks = 0; ks < 2; ++ks) {
      v16bf afr[2], bfr[4];
#pragma unroll
      for (int mt = 0; mt < 2; ++mt)
        afr[mt] = load_frag(bA, wm * 32 + mt * 16 + r, ks, hi);
#pragma unroll
      for (int nt = 0; nt < 4; ++nt)
        bfr[nt] = load_frag(bB, wn * 64 + nt * 16 + r, ks, hi);
#pragma unroll
      for (int mt = 0; mt < 2; ++mt)
#pragma unroll
        for (int nt = 0; nt < 4; ++nt)
          acc[mt][nt] = __builtin_amdgcn_wmma_f32_16x16x32_bf16(
              false, afr[mt], false, bfr[nt], (short)0, acc[mt][nt], false,
              false);
    }
  };

  const int kSteps = K >> 6;
  stage_tiles(A, Bm, lA[0], lB[0], m0, n0, 0, K, tid);
  for (int kb = 0; kb < kSteps - 1; ++kb) {
    __syncthreads();   // buf[(kb+1)&1] consumers (stage kb-1) are done
    stage_tiles(A, Bm, lA[(kb + 1) & 1], lB[(kb + 1) & 1], m0, n0,
                (kb + 1) << 6, K, tid);
    PLE_WAIT8();       // this wave's stage-kb transfers complete
    __syncthreads();   // stage kb visible to all waves
    compute(kb);
  }
  PLE_WAIT0();
  __syncthreads();
  compute(kSteps - 1);

  // epilogue: C layout row = hi*8 + v, col = lane&15
#pragma unroll
  for (int mt = 0; mt < 2; ++mt)
#pragma unroll
    for (int nt = 0; nt < 4; ++nt) {
      int col = n0 + wn * 64 + nt * 16 + r;
      float bv = bias ? bias[col] : 0.0f;
#pragma unroll
      for (int v = 0; v < 8; ++v) {
        int row = m0 + wm * 32 + mt * 16 + hi * 8 + v;
        float val = acc[mt][nt][v] + bv;
        if (GELU) val = gelu_exact(val);
        if (OUT_BF16)
          ((u16*)Cout)[(size_t)row * N + col] = f2bf(val);
        else
          ((float*)Cout)[(size_t)row * N + col] = val;
      }
    }
}

// ---------------------------------------------------------------- LayerNorm
__global__ void __launch_bounds__(256) ln_shared_kernel(
    const float* __restrict__ xs, float* __restrict__ stacked,
    const float* __restrict__ sg, const float* __restrict__ sbt) {
  int gw = (blockIdx.x * 256 + threadIdx.x) >> 5;  // over ES*B
  int lane = threadIdx.x & 31;
  int e = gw >> 12, b = gw & (Bt - 1);
  const float* xrow = xs + (size_t)b * D;
  float* y0 = stacked + (((size_t)e * Bt) + b) * D;   // t=0 slot n=e
  float v[16];
  float s = 0.f;
#pragma unroll
  for (int i = 0; i < 16; ++i) {
    int d = lane + 32 * i;
    v[i] = xrow[d] + y0[d];
    s += v[i];
  }
  float mu = wave_sum(s) * (1.0f / (float)D);
  float q = 0.f;
#pragma unroll
  for (int i = 0; i < 16; ++i) { float dd = v[i] - mu; q += dd * dd; }
  float inv = rsqrtf(wave_sum(q) * (1.0f / (float)D) + EPS);
  float rr[16];
#pragma unroll
  for (int i = 0; i < 16; ++i) {
    int d = lane + 32 * i;
    rr[i] = (v[i] - mu) * inv * sg[(size_t)e * D + d] + sbt[(size_t)e * D + d];
  }
#pragma unroll
  for (int t = 0; t < T; ++t) {
    float* o = stacked + (((size_t)(t * NEXP + e) * Bt) + b) * D;
#pragma unroll
    for (int i = 0; i < 16; ++i) o[lane + 32 * i] = rr[i];
  }
}

__global__ void __launch_bounds__(256) ln_task_kernel(
    const float* __restrict__ x, float* __restrict__ stacked,
    const float* __restrict__ tg, const float* __restrict__ tbt) {
  int gw = (blockIdx.x * 256 + threadIdx.x) >> 5;  // over T*ET*B
  int lane = threadIdx.x & 31;
  int t = gw >> 14, e = (gw >> 12) & 3, b = gw & (Bt - 1);
  const float* xrow = x + ((size_t)t * Bt + b) * D;
  float* y = stacked + (((size_t)(t * NEXP + ES + e) * Bt) + b) * D;
  const float* g = tg + (size_t)(t * ET + e) * D;
  const float* bb = tbt + (size_t)(t * ET + e) * D;
  float v[16];
  float s = 0.f;
#pragma unroll
  for (int i = 0; i < 16; ++i) {
    int d = lane + 32 * i;
    v[i] = xrow[d] + y[d];
    s += v[i];
  }
  float mu = wave_sum(s) * (1.0f / (float)D);
  float q = 0.f;
#pragma unroll
  for (int i = 0; i < 16; ++i) { float dd = v[i] - mu; q += dd * dd; }
  float inv = rsqrtf(wave_sum(q) * (1.0f / (float)D) + EPS);
#pragma unroll
  for (int i = 0; i < 16; ++i) {
    int d = lane + 32 * i;
    y[d] = (v[i] - mu) * inv * g[d] + bb[d];
  }
}

// ---------------------------------------------------------------- gate
__global__ void __launch_bounds__(256) gate_kernel(
    const float* __restrict__ p, const float* __restrict__ stacked,
    float* __restrict__ gated, float* __restrict__ wout) {
  int gw = (blockIdx.x * 256 + threadIdx.x) >> 5;  // over T*B
  int lane = threadIdx.x & 31;
  int t = gw >> 12, b = gw & (Bt - 1);
  const float* prow = p + ((size_t)t * Bt + b) * D;
  float pr[16];
#pragma unroll
  for (int i = 0; i < 16; ++i) pr[i] = prow[lane + 32 * i];

  float st[NEXP][16];
  float sc[NEXP];
#pragma unroll
  for (int n = 0; n < NEXP; ++n) {
    const float* srow = stacked + (((size_t)(t * NEXP + n) * Bt) + b) * D;
    float a = 0.f;
#pragma unroll
    for (int i = 0; i < 16; ++i) {
      st[n][i] = srow[lane + 32 * i];
      a += st[n][i] * pr[i];
    }
    sc[n] = wave_sum(a) * INV_SCALE;
  }
  float mx = sc[0];
#pragma unroll
  for (int n = 1; n < NEXP; ++n) mx = fmaxf(mx, sc[n]);
  float ssum = 0.f;
#pragma unroll
  for (int n = 0; n < NEXP; ++n) { sc[n] = expf(sc[n] - mx); ssum += sc[n]; }
  float invs = 1.0f / ssum;
#pragma unroll
  for (int n = 0; n < NEXP; ++n) sc[n] *= invs;
  if (lane < NEXP) wout[((size_t)t * Bt + b) * NEXP + lane] = sc[lane];
#pragma unroll
  for (int i = 0; i < 16; ++i) {
    float g = 0.f;
#pragma unroll
    for (int n = 0; n < NEXP; ++n) g += sc[n] * st[n][i];
    gated[((size_t)t * Bt + b) * D + lane + 32 * i] = g;
  }
}

// ---------------------------------------------------------------- host
static inline void* ws_take(char*& cur, size_t bytes) {
  void* p = (void*)cur;
  cur += (bytes + 255) & ~(size_t)255;
  return p;
}

extern "C" void kernel_launch(void* const* d_in, const int* in_sizes, int n_in,
                              void* d_out, int out_size, void* d_ws,
                              size_t ws_size, hipStream_t stream) {
  (void)in_sizes; (void)n_in; (void)out_size; (void)ws_size;
  const float* x   = (const float*)d_in[0];
  const float* sw1 = (const float*)d_in[1];
  const float* sb1 = (const float*)d_in[2];
  const float* sw2 = (const float*)d_in[3];
  const float* sb2 = (const float*)d_in[4];
  const float* sg  = (const float*)d_in[5];
  const float* sbt = (const float*)d_in[6];
  const float* tw1 = (const float*)d_in[7];
  const float* tb1 = (const float*)d_in[8];
  const float* tw2 = (const float*)d_in[9];
  const float* tb2 = (const float*)d_in[10];
  const float* tg  = (const float*)d_in[11];
  const float* tbt = (const float*)d_in[12];
  const float* qw  = (const float*)d_in[13];
  const float* qb  = (const float*)d_in[14];
  const float* kw  = (const float*)d_in[15];

  float* gated   = (float*)d_out;
  float* wout    = gated + (size_t)T * Bt * D;
  float* stacked = wout + (size_t)T * Bt * NEXP;

  char* cur = (char*)d_ws;
  float* xs_f   = (float*)ws_take(cur, (size_t)Bt * D * 4);
  u16*   xs_b   = (u16*)  ws_take(cur, (size_t)Bt * D * 2);
  u16*   x_b    = (u16*)  ws_take(cur, (size_t)T * Bt * D * 2);
  u16*   sw1_bt = (u16*)  ws_take(cur, (size_t)ES * D * H * 2);   // [H,D] per e
  u16*   sw2_bt = (u16*)  ws_take(cur, (size_t)ES * H * D * 2);   // [D,H] per e
  u16*   tw1_bt = (u16*)  ws_take(cur, (size_t)T * ET * D * H * 2);
  u16*   tw2_bt = (u16*)  ws_take(cur, (size_t)T * ET * H * D * 2);
  u16*   qw_b   = (u16*)  ws_take(cur, (size_t)T * D * D * 2);    // [f,e] rows
  u16*   kw_b   = (u16*)  ws_take(cur, (size_t)T * D * D * 2);    // [d,e] rows
  u16*   G_b    = (u16*)  ws_take(cur, (size_t)T * D * D * 2);    // G'=kw@qw^T
  float* cb     = (float*)ws_take(cur, (size_t)T * D * 4);
  u16*   h_b    = (u16*)  ws_take(cur, (size_t)Bt * H * 2);
  float* p_f    = (float*)ws_take(cur, (size_t)T * Bt * D * 4);

  // ---- prep: mean/bf16 casts, transposed bf16 weights, folded gate bias
  prep_x_kernel<<<(Bt * D) / 256, 256, 0, stream>>>(x, xs_f, xs_b, x_b);
  cvt_t_kernel<<<dim3(H / 32, D / 32, ES), 256, 0, stream>>>(sw1, sw1_bt, D, H);
  cvt_t_kernel<<<dim3(D / 32, H / 32, ES), 256, 0, stream>>>(sw2, sw2_bt, H, D);
  cvt_t_kernel<<<dim3(H / 32, D / 32, T * ET), 256, 0, stream>>>(tw1, tw1_bt,
                                                                 D, H);
  cvt_t_kernel<<<dim3(D / 32, H / 32, T * ET), 256, 0, stream>>>(tw2, tw2_bt,
                                                                 H, D);
  cvt_bf16_kernel<<<(T * D * D) / 256, 256, 0, stream>>>(qw, qw_b);
  cvt_bf16_kernel<<<(T * D * D) / 256, 256, 0, stream>>>(kw, kw_b);
  cbias_kernel<<<(T * D) / 256, 256, 0, stream>>>(qb, kw, cb);

  // ---- G'[t] = kw[t] @ qw[t]^T  (so p = x @ G'^T uses B stored [N,K])
  for (int t = 0; t < T; ++t) {
    gemm_bf16_kernel<false, true><<<dim3(D / 128, D / 128), 256, 0, stream>>>(
        kw_b + (size_t)t * D * D, qw_b + (size_t)t * D * D, nullptr,
        (void*)(G_b + (size_t)t * D * D), D, D, D);
  }

  // ---- shared experts
  for (int e = 0; e < ES; ++e) {
    gemm_bf16_kernel<true, true><<<dim3(H / 128, Bt / 128), 256, 0, stream>>>(
        xs_b, sw1_bt + (size_t)e * D * H, sb1 + (size_t)e * H, (void*)h_b, Bt,
        H, D);
    gemm_bf16_kernel<false, false><<<dim3(D / 128, Bt / 128), 256, 0,
                                     stream>>>(
        h_b, sw2_bt + (size_t)e * H * D, sb2 + (size_t)e * D,
        (void*)(stacked + ((size_t)e * Bt) * D), Bt, D, H);
  }
  ln_shared_kernel<<<(ES * Bt * 32) / 256, 256, 0, stream>>>(xs_f, stacked, sg,
                                                             sbt);

  // ---- task experts
  for (int t = 0; t < T; ++t)
    for (int e = 0; e < ET; ++e) {
      int be = t * ET + e;
      gemm_bf16_kernel<true, true><<<dim3(H / 128, Bt / 128), 256, 0,
                                     stream>>>(
          x_b + (size_t)t * Bt * D, tw1_bt + (size_t)be * D * H,
          tb1 + (size_t)be * H, (void*)h_b, Bt, H, D);
      gemm_bf16_kernel<false, false><<<dim3(D / 128, Bt / 128), 256, 0,
                                       stream>>>(
          h_b, tw2_bt + (size_t)be * H * D, tb2 + (size_t)be * D,
          (void*)(stacked + ((size_t)(t * NEXP + ES + e) * Bt) * D), Bt, D, H);
    }
  ln_task_kernel<<<(T * ET * Bt * 32) / 256, 256, 0, stream>>>(x, stacked, tg,
                                                               tbt);

  // ---- p[t] = x[t] @ G'[t]^T + cb[t]
  for (int t = 0; t < T; ++t) {
    gemm_bf16_kernel<false, false><<<dim3(D / 128, Bt / 128), 256, 0,
                                     stream>>>(
        x_b + (size_t)t * Bt * D, G_b + (size_t)t * D * D, cb + (size_t)t * D,
        (void*)(p_f + (size_t)t * Bt * D), Bt, D, D);
  }

  // ---- softmax gate + mix
  gate_kernel<<<(T * Bt * 32) / 256, 256, 0, stream>>>(p_f, stacked, gated,
                                                       wout);
}
